// FastKANLayer_17025250362073
// MI455X (gfx1250) — compile-verified, matrix operand order checked by reading, not run on hardware
//
#include <hip/hip_runtime.h>

// ---------------------------------------------------------------------------
// FastKAN layer for MI455X (gfx1250, wave32, WMMA).
// GEMMs: v_wmma_f32_16x16x32_bf16, f32 accumulate.
// Fragments: 2x b128 vector loads per lane (contiguous-K runs of the 16-bit
// A/B VGPR layout). Q tiles streamed from HBM with global_prefetch ahead of
// the WMMA chain.
// ---------------------------------------------------------------------------

typedef __attribute__((ext_vector_type(16))) __bf16 v16bf;
typedef __attribute__((ext_vector_type(8)))  __bf16 v8bf;
typedef __attribute__((ext_vector_type(8)))  float  v8f;

#define B_    8
#define IN_N  8
#define OUT_N 8
#define SEQ   256
#define FEAT  64
#define NC    99
#define KSZ   15
#define DKQ   256   // (IN_N/2) * FEAT

// ---------------- wave32 reductions ----------------
__device__ __forceinline__ float wred_sum(float v) {
#pragma unroll
  for (int m = 16; m >= 1; m >>= 1) v += __shfl_xor(v, m, 32);
  return v;
}
__device__ __forceinline__ float wred_max(float v) {
#pragma unroll
  for (int m = 16; m >= 1; m >>= 1) v = fmaxf(v, __shfl_xor(v, m, 32));
  return v;
}

// ---------------- WMMA fragment load (16-bit A / B layout) -----------------
// 16 rows x 32 K tile, element (r,k) at base[r*stride + k].
// Lane L (r=L&15, half=L>>4): vector elems 0..7  = base[r*stride+half*8+0..7]
//                              vector elems 8..15 = base[r*stride+16+half*8+0..7]
// => two contiguous 16-byte loads per lane (ds_load_b128 / global_load_b128).
__device__ __forceinline__ v16bf frag_ld(const __bf16* base, int stride, int lane) {
  int r = lane & 15, half = lane >> 4;
  const __bf16* p = base + r * stride + (half << 3);
  v8bf lo = *(const v8bf*)(p);
  v8bf hi = *(const v8bf*)(p + 16);
  return __builtin_shufflevector(lo, hi, 0, 1, 2, 3, 4, 5, 6, 7,
                                 8, 9, 10, 11, 12, 13, 14, 15);
}

// ---------------- kernel 0: zero energy accumulators -----------------------
__global__ void init_kernel(float* energy) {
  int t = threadIdx.x;
  if (t < B_ * IN_N * OUT_N) energy[t] = 0.0f;
}

// ---------------- kernel 1: per-(b,i,f) mean/var over SEQ ------------------
__global__ __launch_bounds__(256) void stats_kernel(const float* __restrict__ x_in,
                                                    float* __restrict__ mu,
                                                    float* __restrict__ var) {
  int bi = blockIdx.x;                       // b*IN_N + i
  int f = threadIdx.x & 63, g = threadIdx.x >> 6;
  const float* xb = x_in + (size_t)bi * SEQ * FEAT;
  float s = 0.f, s2 = 0.f;
  for (int t = g; t < SEQ; t += 4) {
    float v = xb[t * FEAT + f];
    s += v; s2 += v * v;
  }
  __shared__ float r1[4][64], r2[4][64];
  r1[g][f] = s; r2[g][f] = s2;
  __syncthreads();
  if (g == 0) {
    s  = r1[0][f] + r1[1][f] + r1[2][f] + r1[3][f];
    s2 = r2[0][f] + r2[1][f] + r2[2][f] + r2[3][f];
    float m = s * (1.0f / SEQ);
    mu[bi * 64 + f]  = m;
    var[bi * 64 + f] = s2 * (1.0f / SEQ) - m * m;
  }
}

// ---------------- kernel 2: compact-support spline + edges + energy --------
// Cubic B-spline on a uniform grid has only 4 non-zero taps -> replaces the
// dense 99-wide contraction.
__global__ __launch_bounds__(256) void spline_kernel(const float* __restrict__ x_in,
                                                     const float* __restrict__ mu,
                                                     const float* __restrict__ var,
                                                     const float* __restrict__ sw,
                                                     const float* __restrict__ omiga,
                                                     float* __restrict__ act,
                                                     float* __restrict__ energy) {
  int blk   = blockIdx.x;                    // (b*IN_N+i)*64 + stile
  int stile = blk & 63;
  int bi    = blk >> 6;
  int i     = bi & 7;
  int f  = threadIdx.x & 63, sl = threadIdx.x >> 6;
  int s  = stile * 4 + sl;
  float x = x_in[(size_t)bi * SEQ * FEAT + (size_t)s * FEAT + f];
  float m = mu[bi * 64 + f], v = var[bi * 64 + f];
  float xn = (x - m) * rsqrtf(v + 1e-5f);
  xn = fminf(fmaxf(xn * 0.5f, -0.99f), 0.99f);

  const float h = 2.0f / (float)(NC - 1);
  int k0 = (int)floorf((xn + 1.0f) / h);
  float bas[4]; int cc[4];
#pragma unroll
  for (int tp = 0; tp < 4; ++tp) {
    int c = k0 - 1 + tp; cc[tp] = c;
    float g = -1.0f + (float)c * h;
    float dist = fabsf(xn - g) / h;
    float a2 = fmaxf(2.0f - dist, 0.0f);
    float a1 = fmaxf(1.0f - dist, 0.0f);
    float b = a2 * a2 * a2 * (1.0f / 6.0f) - a1 * a1 * a1 * (4.0f / 6.0f);
    bas[tp] = (c < 0 || c >= NC) ? 0.0f : b;
  }

  __shared__ float part[OUT_N][8];
  int wv = threadIdx.x >> 5, ln = threadIdx.x & 31;
#pragma unroll
  for (int o = 0; o < OUT_N; ++o) {
    const float* w = sw + ((size_t)i * OUT_N + o) * NC;
    float acc = 0.f;
#pragma unroll
    for (int tp = 0; tp < 4; ++tp) {
      int c = min(max(cc[tp], 0), NC - 1);
      acc += bas[tp] * w[c];
    }
    float edge = acc + fabsf(omiga[i * OUT_N + o]) * x;
    act[(((size_t)bi * OUT_N + o) * SEQ + s) * FEAT + f] = edge;
    float t = wred_sum(edge * edge);
    if (ln == 0) part[o][wv] = t;
  }
  __syncthreads();
  if (threadIdx.x < OUT_N) {
    float t = 0.f;
#pragma unroll
    for (int w = 0; w < 8; ++w) t += part[threadIdx.x][w];
    atomicAdd(&energy[bi * OUT_N + threadIdx.x], t);
  }
}

// ---------------- kernel 3: energy gate ------------------------------------
__global__ void gate_kernel(const float* __restrict__ energy,
                            const float* __restrict__ tau,
                            const float* __restrict__ temperature,
                            float* __restrict__ coefwp,
                            float* __restrict__ cmask) {
  int t = threadIdx.x;                       // (b*IN_N+i)*OUT_N + j
  if (t >= B_ * IN_N * OUT_N) return;
  int j = t & 7, bi = t >> 3, i = bi & 7;
  float tv = fabsf(temperature[0]) + 1e-4f;  // sqrt(SEQ/256) == 1
  float em = energy[t] * (1.0f / (float)(SEQ * FEAT));
  float es = sqrtf(em + 1e-8f);
  float tj = fabsf(tau[i * OUT_N + j]);
  float mask = 1.0f / (1.0f + expf(-(es - tj) / tv));
  coefwp[t] = (es / (tj + 1e-8f)) * mask;
  cmask[t]  = mask;
}

// ---------------- kernel 4: build K/Q rows, LayerNorm, store bf16 ----------
__global__ __launch_bounds__(256) void kq_kernel(const float* __restrict__ proj,
                                                 const float* __restrict__ coefwp,
                                                 __bf16* __restrict__ Kmat,
                                                 __bf16* __restrict__ Qmat) {
  int blk = blockIdx.x;                      // (b*OUT_N+j)*SEQ + t
  int t = blk & 255, bj = blk >> 8, j = bj & 7, b = bj >> 3;
  int d = threadIdx.x;
  int ihalf = d >> 6, f = d & 63;
  int wv = d >> 5, ln = d & 31;
  __shared__ float part[8];
#pragma unroll
  for (int q = 0; q < 2; ++q) {
    int i = 2 * ihalf + q;
    float v = proj[(((size_t)b * IN_N + i) * SEQ + t) * FEAT + f] *
              coefwp[(b * IN_N + i) * OUT_N + j];
    float w = wred_sum(v);
    if (ln == 0) part[wv] = w;
    __syncthreads();
    float mean = 0.f;
#pragma unroll
    for (int k = 0; k < 8; ++k) mean += part[k];
    mean *= (1.0f / 256.0f);
    __syncthreads();
    w = wred_sum(v * v);
    if (ln == 0) part[wv] = w;
    __syncthreads();
    float msq = 0.f;
#pragma unroll
    for (int k = 0; k < 8; ++k) msq += part[k];
    msq *= (1.0f / 256.0f);
    float xn = (v - mean) * rsqrtf(msq - mean * mean + 1e-5f);
    __bf16* dst = q ? Qmat : Kmat;
    dst[((size_t)bj * SEQ + t) * DKQ + d] = (__bf16)xn;
    __syncthreads();
  }
}

// ---------------- kernel 5: combined -> LN -> x' = xn @ W2 + bs (WMMA) -----
__global__ __launch_bounds__(32) void gemm2_kernel(const float* __restrict__ act,
                                                   const float* __restrict__ cmask,
                                                   const float* __restrict__ ln_w,
                                                   const float* __restrict__ ln_b,
                                                   const float* __restrict__ W2s,
                                                   const float* __restrict__ bs,
                                                   float* __restrict__ comb_out,
                                                   float* __restrict__ xp_out,
                                                   float* __restrict__ projs_out) {
  int blk = blockIdx.x;                      // (b*OUT_N+j)*16 + mtile
  int mtile = blk & 15, bj = blk >> 4, j = bj & 7, b = bj >> 3;
  int lane = threadIdx.x;
  int m0 = mtile * 16;
  __shared__ __align__(16) __bf16 xn_s[16 * FEAT];     // A: [r][k]
  __shared__ __align__(16) __bf16 w2t_s[FEAT * FEAT];  // B^T: [n][k] = W2[k][n]

  float cm[IN_N];
#pragma unroll
  for (int i = 0; i < IN_N; ++i) cm[i] = cmask[(b * IN_N + i) * OUT_N + j];

  for (int r = 0; r < 16; ++r) {
    int s = m0 + r;
    int f0 = lane, f1 = lane + 32;
    float v0 = 0.f, v1 = 0.f;
#pragma unroll
    for (int i = 0; i < IN_N; ++i) {
      const float* e = act + (((size_t)(b * IN_N + i) * OUT_N + j) * SEQ + s) * FEAT;
      // prefetch next row of this edge stream while reducing the current one
      if (r < 15) __builtin_prefetch(e + FEAT, 0, 1);
      v0 += e[f0] * cm[i];
      v1 += e[f1] * cm[i];
    }
    comb_out[((size_t)bj * SEQ + s) * FEAT + f0] = v0;
    comb_out[((size_t)bj * SEQ + s) * FEAT + f1] = v1;
    float mean = wred_sum(v0 + v1) * (1.0f / 64.0f);
    float msq  = wred_sum(v0 * v0 + v1 * v1) * (1.0f / 64.0f);
    float inv  = rsqrtf(msq - mean * mean + 1e-5f);
    xn_s[r * FEAT + f0] = (__bf16)((v0 - mean) * inv * ln_w[j * FEAT + f0] + ln_b[j * FEAT + f0]);
    xn_s[r * FEAT + f1] = (__bf16)((v1 - mean) * inv * ln_w[j * FEAT + f1] + ln_b[j * FEAT + f1]);
  }
  // transpose W2 into LDS: w2t[n][k] = W2[k][n]
  const float* w2 = W2s + (size_t)j * FEAT * FEAT;
  for (int idx = lane; idx < FEAT * FEAT; idx += 32) {
    int k = idx >> 6, n = idx & 63;
    w2t_s[n * FEAT + k] = (__bf16)w2[idx];
  }
  __syncthreads();

#pragma unroll
  for (int nt = 0; nt < 4; ++nt) {
    v8f acc = {};
#pragma unroll
    for (int ks = 0; ks < 2; ++ks) {
      v16bf a  = frag_ld(xn_s + ks * 32, FEAT, lane);
      v16bf bm = frag_ld(w2t_s + (nt * 16) * FEAT + ks * 32, FEAT, lane);
      acc = __builtin_amdgcn_wmma_f32_16x16x32_bf16(false, a, false, bm,
                                                    (short)0, acc, false, false);
    }
#pragma unroll
    for (int v = 0; v < 8; ++v) {
      int m = ((lane >> 4) << 3) | v;
      int n = lane & 15;
      int s = m0 + m, f = nt * 16 + n;
      float val = acc[v] + bs[((size_t)j * SEQ + s) * FEAT + f];
      size_t o = ((size_t)bj * SEQ + s) * FEAT + f;
      xp_out[o]    = val;
      projs_out[o] = val;
    }
  }
}

// ---------------- kernel 6: fused attention (scores, softmax, spatial) -----
__global__ __launch_bounds__(32) void attn_kernel(const __bf16* __restrict__ Kmat,
                                                  const __bf16* __restrict__ Qmat,
                                                  const float* __restrict__ xp,
                                                  const float* __restrict__ comb,
                                                  const float* __restrict__ convk,
                                                  const float* __restrict__ temperature,
                                                  const float* __restrict__ alphas,
                                                  const float* __restrict__ betas,
                                                  const float* __restrict__ thetas,
                                                  const float* __restrict__ gammas,
                                                  float* __restrict__ d_outs) {
  int blk = blockIdx.x;                      // (b*OUT_N+j)*16 + mtile
  int mtile = blk & 15, bj = blk >> 4, j = bj & 7;
  int lane = threadIdx.x;
  int m0 = mtile * 16;

  __shared__ __align__(16) __bf16 k_s[16 * DKQ];    // 8 KB: K tile, then probs
  __shared__ __align__(16) float  sc_s[16 * SEQ];   // 16 KB: raw scores
  __shared__ __align__(16) __bf16 xp_t[FEAT * SEQ]; // 32 KB: x' transposed [f][t]
  __shared__ __align__(16) float  ck_s[FEAT * KSZ]; // ~4 KB: conv taps

  const __bf16* Qbase = Qmat + (size_t)bj * SEQ * DKQ;

  // prefetch the first two Q tiles into cache before anything else
  __builtin_prefetch(Qbase + (size_t)lane * DKQ, 0, 1);
  __builtin_prefetch(Qbase + (size_t)(16 + lane) * DKQ, 0, 1);

  // K tile fill: b128 copies
  {
    const uint4* src = (const uint4*)(Kmat + ((size_t)bj * SEQ + m0) * DKQ);
    uint4* dst = (uint4*)k_s;
    for (int idx = lane; idx < 16 * DKQ / 8; idx += 32) dst[idx] = src[idx];
  }
  // x' transposed into LDS (bf16)
  {
    const float* xpb = xp + (size_t)bj * SEQ * FEAT;
    for (int idx = lane; idx < FEAT * SEQ; idx += 32) {
      int f = idx >> 8, t = idx & 255;
      xp_t[idx] = (__bf16)xpb[t * FEAT + f];
    }
  }
  {
    const float* ckb = convk + (size_t)j * FEAT * KSZ;
    for (int idx = lane; idx < FEAT * KSZ; idx += 32) ck_s[idx] = ckb[idx];
  }
  __syncthreads();

  // raw = K . Q^T  (16 x 256): A from LDS, B streamed from HBM (b128 loads),
  // tile two ahead prefetched behind the current WMMA chain.
  for (int tt = 0; tt < 16; ++tt) {
    if (tt < 14)  // prefetch Q rows two tiles ahead (one row per lane, 2 lines)
      __builtin_prefetch(Qbase + (size_t)((tt + 2) * 16 + (lane & 15)) * DKQ +
                             ((lane >> 4) << 7), 0, 1);
    v8f acc = {};
#pragma unroll
    for (int ks = 0; ks < 8; ++ks) {
      v16bf a  = frag_ld(k_s + ks * 32, DKQ, lane);
      v16bf bm = frag_ld(Qbase + (size_t)(tt * 16) * DKQ + ks * 32, DKQ, lane);
      acc = __builtin_amdgcn_wmma_f32_16x16x32_bf16(false, a, false, bm,
                                                    (short)0, acc, false, false);
    }
#pragma unroll
    for (int v = 0; v < 8; ++v) {
      int m = ((lane >> 4) << 3) | v;
      int n = lane & 15;
      sc_s[m * SEQ + tt * 16 + n] = acc[v];
    }
  }
  __syncthreads();   // K tile dead from here; k_s is reused for bf16 probs

  // row softmax of raw/(16*t_val); write probabilities as bf16 into k_s
  float tv = fabsf(temperature[0]) + 1e-4f;
  float scale = 1.0f / (16.0f * tv);
  for (int r = 0; r < 16; ++r) {
    float mx = -3.4e38f;
    for (int c = lane; c < SEQ; c += 32) mx = fmaxf(mx, sc_s[r * SEQ + c] * scale);
    mx = wred_max(mx);
    float sum = 0.f;
    for (int c = lane; c < SEQ; c += 32) {
      float e = expf(sc_s[r * SEQ + c] * scale - mx);
      sc_s[r * SEQ + c] = e;
      sum += e;
    }
    sum = wred_sum(sum);
    float inv = 1.0f / sum;
    for (int c = lane; c < SEQ; c += 32)
      k_s[r * SEQ + c] = (__bf16)(sc_s[r * SEQ + c] * inv);
  }
  __syncthreads();

  // spatial = |beta| * W1 @ x' + |alpha| * x' + |theta| * conv + gamma * combined
  float ab = fabsf(alphas[j]), bb = fabsf(betas[j]);
  float th = fabsf(thetas[j]), gm = gammas[j];
  const float* combb = comb + (size_t)bj * SEQ * FEAT;
#pragma unroll
  for (int nt = 0; nt < 4; ++nt) {
    v8f acc = {};
#pragma unroll
    for (int ks = 0; ks < 8; ++ks) {
      v16bf a  = frag_ld(k_s + ks * 32, SEQ, lane);                    // probs bf16
      v16bf bm = frag_ld(xp_t + (nt * 16) * SEQ + ks * 32, SEQ, lane); // x'^T
      acc = __builtin_amdgcn_wmma_f32_16x16x32_bf16(false, a, false, bm,
                                                    (short)0, acc, false, false);
    }
#pragma unroll
    for (int v = 0; v < 8; ++v) {
      int m = ((lane >> 4) << 3) | v;
      int n = lane & 15;
      int s = m0 + m, f = nt * 16 + n;
      float xval = (float)xp_t[f * SEQ + s];
      float w3 = 0.f;
#pragma unroll
      for (int k = 0; k < KSZ; ++k) {
        int sp = s + k - KSZ / 2;
        if (sp >= 0 && sp < SEQ) w3 += ck_s[f * KSZ + k] * (float)xp_t[f * SEQ + sp];
      }
      float val = bb * acc[v] + ab * xval + th * w3 + gm * combb[s * FEAT + f];
      d_outs[((size_t)bj * SEQ + s) * FEAT + f] = val;
    }
  }
}

// ---------------------------------------------------------------------------
extern "C" void kernel_launch(void* const* d_in, const int* in_sizes, int n_in,
                              void* d_out, int out_size, void* d_ws, size_t ws_size,
                              hipStream_t stream) {
  (void)in_sizes; (void)n_in; (void)out_size; (void)ws_size;
  const float* x_in   = (const float*)d_in[0];
  const float* proj   = (const float*)d_in[1];
  /* d_in[2] = grid (uniform linspace; evaluated analytically) */
  const float* sw     = (const float*)d_in[3];
  const float* tau    = (const float*)d_in[4];
  const float* temp   = (const float*)d_in[5];
  const float* omiga  = (const float*)d_in[6];
  const float* W2s    = (const float*)d_in[7];
  const float* bs     = (const float*)d_in[8];
  const float* ln_w   = (const float*)d_in[9];
  const float* ln_b   = (const float*)d_in[10];
  const float* alphas = (const float*)d_in[11];
  const float* betas  = (const float*)d_in[12];
  const float* thetas = (const float*)d_in[13];
  const float* gammas = (const float*)d_in[14];
  const float* convk  = (const float*)d_in[15];

  // workspace layout
  float* F      = (float*)d_ws;
  float* mu     = F;                 // 4096
  float* var    = F + 4096;          // 4096
  float* energy = F + 8192;          // 512
  float* coefwp = F + 8704;          // 512
  float* cmask  = F + 9216;          // 512
  float* act    = F + 9728;          // 8,388,608
  float* comb   = act  + (size_t)B_ * IN_N * OUT_N * SEQ * FEAT;   // 1,048,576
  float* xp     = comb + (size_t)B_ * OUT_N * SEQ * FEAT;          // 1,048,576
  __bf16* Kmat  = (__bf16*)(xp + (size_t)B_ * OUT_N * SEQ * FEAT);
  __bf16* Qmat  = Kmat + (size_t)B_ * OUT_N * SEQ * DKQ;

  float* outs  = (float*)d_out;
  float* projs = outs + (size_t)B_ * OUT_N * SEQ * FEAT;

  init_kernel <<<1, 512, 0, stream>>>(energy);
  stats_kernel<<<B_ * IN_N, 256, 0, stream>>>(x_in, mu, var);
  spline_kernel<<<B_ * IN_N * 64, 256, 0, stream>>>(x_in, mu, var, sw, omiga, act, energy);
  gate_kernel <<<1, 512, 0, stream>>>(energy, tau, temp, coefwp, cmask);
  kq_kernel   <<<B_ * OUT_N * SEQ, 256, 0, stream>>>(proj, coefwp, Kmat, Qmat);
  gemm2_kernel<<<B_ * OUT_N * 16, 32, 0, stream>>>(act, cmask, ln_w, ln_b, W2s, bs,
                                                   comb, xp, projs);
  attn_kernel <<<B_ * OUT_N * 16, 32, 0, stream>>>(Kmat, Qmat, xp, comb, convk, temp,
                                                   alphas, betas, thetas, gammas, outs);
}
